// Int8OPTAttention_1752346657455
// MI455X (gfx1250) — 8-bit matrix operand order repaired and checked by reading, compiled, not run on hardware
//
#include <hip/hip_runtime.h>
#include <stdint.h>


#define B_ 2
#define T_ 2048
#define E_ 4096
#define H_ 32
#define D_ 128

typedef int v8i __attribute__((ext_vector_type(8)));

static __device__ __forceinline__ v8i v8zero() {
  v8i r = {0, 0, 0, 0, 0, 0, 0, 0};
  return r;
}

static __device__ __forceinline__ v8i wmma_iu8(v8i a, v8i b, v8i c) {
  return __builtin_amdgcn_wmma_i32_16x16x64_iu8(true, a, true, b, c, false, false);
}

static __device__ __forceinline__ v8i pack8(uint2 d0, uint2 d1, uint2 d2, uint2 d3) {
  v8i r;
  r[0] = (int)d0.x; r[1] = (int)d0.y; r[2] = (int)d1.x; r[3] = (int)d1.y;
  r[4] = (int)d2.x; r[5] = (int)d2.y; r[6] = (int)d3.x; r[7] = (int)d3.y;
  return r;
}
static __device__ __forceinline__ v8i pack8q(uint4 d0, uint4 d1) {
  v8i r;
  r[0] = (int)d0.x; r[1] = (int)d0.y; r[2] = (int)d0.z; r[3] = (int)d0.w;
  r[4] = (int)d1.x; r[5] = (int)d1.y; r[6] = (int)d1.z; r[7] = (int)d1.w;
  return r;
}

static __device__ __forceinline__ void async_ld16(const int8_t* g, int8_t* lds) {
  asm volatile("global_load_async_to_lds_b128 %0, %1, off"
               :
               : "v"((uint32_t)(uintptr_t)lds), "v"((uint64_t)(uintptr_t)g)
               : "memory");
}
#define ASYNC_WAIT_4() asm volatile("s_wait_asynccnt 0x4" ::: "memory")
#define ASYNC_WAIT_0() asm volatile("s_wait_asynccnt 0x0" ::: "memory")

static __device__ __forceinline__ v8i a_frag_lds(const int8_t* tile, int rbase, int lane) {
  const int m = lane & 15;
  const int8_t* row = tile + (rbase + m) * 64 + ((lane >> 4) << 4);
  return pack8q(*(const uint4*)(row), *(const uint4*)(row + 32));
}

static __device__ __forceinline__ v8i b_frag_lds(const int8_t* tile, int nbase, int lane) {
  const int n = lane & 15;
  const int8_t* row = tile + (nbase + n) * 64 + ((lane >> 4) << 4);
  return pack8q(*(const uint4*)(row), *(const uint4*)(row + 32));
}

static __device__ __forceinline__ int8_t sat8(float y) {
  int q = __float2int_rn(y);
  q = q < -128 ? -128 : (q > 127 ? 127 : q);
  return (int8_t)q;
}

static __device__ __forceinline__ void stage_tile_async(
    const int8_t* __restrict__ X, const int8_t* __restrict__ W,
    int8_t* als, int8_t* bls, int m0, int n0, int kt, int tid) {
  const int r = tid >> 2, off = (tid & 3) << 4;
  async_ld16(&X[(size_t)(m0 + r) * E_ + kt + off], &als[r * 64 + off]);
  async_ld16(&X[(size_t)(m0 + r + 64) * E_ + kt + off], &als[(r + 64) * 64 + off]);
  async_ld16(&W[(size_t)(n0 + r) * E_ + kt + off], &bls[r * 64 + off]);
  async_ld16(&W[(size_t)(n0 + r + 64) * E_ + kt + off], &bls[(r + 64) * 64 + off]);
}

__global__ void k_pack_i8(const int* __restrict__ src, int8_t* __restrict__ dst, int n4) {
  int i = blockIdx.x * blockDim.x + threadIdx.x;
  const int stride = gridDim.x * blockDim.x;
  const int4* s4 = (const int4*)src;
  int* d4 = (int*)dst;
  for (; i < n4; i += stride) {
    int4 v = s4[i];
    d4[i] = (v.x & 255) | ((v.y & 255) << 8) | ((v.z & 255) << 16) | (v.w << 24);
  }
}

__global__ __launch_bounds__(256) void k_gemm_w8a8(
    const int8_t* __restrict__ X, const int8_t* __restrict__ W,
    const int8_t* __restrict__ bias, const float* __restrict__ alpha_p,
    int8_t* __restrict__ out, int vt_mode) {
  __shared__ int8_t Als[2][128 * 64];
  __shared__ int8_t Bls[2][128 * 64];
  const int tid = threadIdx.x;
  const int lane = tid & 31, wave = tid >> 5;
  const int wm = wave >> 2, wn = wave & 3;
  const int m0 = blockIdx.y * 128, n0 = blockIdx.x * 128;

  v8i acc[4][2];
#pragma unroll
  for (int i = 0; i < 4; ++i)
#pragma unroll
    for (int j = 0; j < 2; ++j) acc[i][j] = v8zero();

  stage_tile_async(X, W, Als[0], Bls[0], m0, n0, 0, tid);
  int buf = 0;
  for (int kt = 0; kt < E_; kt += 64) {
    if (kt + 64 < E_) {
      stage_tile_async(X, W, Als[buf ^ 1], Bls[buf ^ 1], m0, n0, kt + 64, tid);
      ASYNC_WAIT_4();
    } else {
      ASYNC_WAIT_0();
    }
    __syncthreads();
    v8i bfr0 = b_frag_lds(Bls[buf], wn * 32, lane);
    v8i bfr1 = b_frag_lds(Bls[buf], wn * 32 + 16, lane);
#pragma unroll
    for (int mt = 0; mt < 4; ++mt) {
      v8i afr = a_frag_lds(Als[buf], wm * 64 + mt * 16, lane);
      acc[mt][0] = wmma_iu8(afr, bfr0, acc[mt][0]);
      acc[mt][1] = wmma_iu8(afr, bfr1, acc[mt][1]);
    }
    __syncthreads();
    buf ^= 1;
  }

  const float alpha = *alpha_p;
  const int cl = lane & 15, hi = lane >> 4;
#pragma unroll
  for (int mt = 0; mt < 4; ++mt)
#pragma unroll
    for (int nt = 0; nt < 2; ++nt) {
      const int col = n0 + wn * 32 + nt * 16 + cl;
      const float bv = (float)bias[col];
#pragma unroll
      for (int j = 0; j < 8; ++j) {
        const int row = m0 + wm * 64 + mt * 16 + j + 8 * hi;
        const int8_t q = sat8((float)acc[mt][nt][j] * alpha + bv);
        if (!vt_mode) {
          out[(size_t)row * E_ + col] = q;
        } else {
          const int b = row >> 11, t = row & (T_ - 1);
          const int h = col >> 7, d = col & (D_ - 1);
          out[(((size_t)(b * H_ + h) * D_ + d) << 11) + t] = q;
        }
      }
    }
}

__global__ __launch_bounds__(256) void k_attn(
    const int8_t* __restrict__ Q8, const int8_t* __restrict__ K8,
    const int8_t* __restrict__ VT8, int8_t* __restrict__ CTX8,
    const float* __restrict__ aqk_p, const float* __restrict__ apv_p) {
  __shared__ int8_t Pls[8][16 * 64];
  const int tid = threadIdx.x, lane = tid & 31, wave = tid >> 5;
  const int bh = blockIdx.y, b = bh >> 5, h = bh & (H_ - 1);
  const int r0 = blockIdx.x * 128 + wave * 16;
  const int m = lane & 15, hi = lane >> 4;
  const int hi8 = hi << 3, hi16 = hi << 4;
  const float aqk = *aqk_p, apv = *apv_p;
  int8_t* pbuf = Pls[wave];

  const int8_t* qb = Q8 + (size_t)(b * T_ + r0 + m) * E_ + h * D_ + hi16;
  v8i aQ[2];
#pragma unroll
  for (int kk = 0; kk < 2; ++kk) {
    const int8_t* p = qb + kk * 64;
    aQ[kk] = pack8q(*(const uint4*)(p), *(const uint4*)(p + 32));
  }

  float mrow[8], srow[8];
#pragma unroll
  for (int j = 0; j < 8; ++j) { mrow[j] = -3.0e38f; srow[j] = 0.f; }

  for (int s0 = 0; s0 < T_; s0 += 16) {
    const int8_t* kb = K8 + (size_t)(b * T_ + s0 + m) * E_ + h * D_ + hi16;
    v8i bk0 = pack8q(*(const uint4*)(kb), *(const uint4*)(kb + 32));
    v8i bk1 = pack8q(*(const uint4*)(kb + 64), *(const uint4*)(kb + 96));
    v8i acc = v8zero();
    acc = wmma_iu8(aQ[0], bk0, acc);
    acc = wmma_iu8(aQ[1], bk1, acc);
    const int scol = s0 + m;
#pragma unroll
    for (int j = 0; j < 8; ++j) {
      const int row = r0 + j + 8 * hi;
      const float l = aqk * (float)acc[j] + (scol <= row ? 0.f : -1e4f);
      const float mn = fmaxf(mrow[j], l);
      srow[j] = srow[j] * __expf(mrow[j] - mn) + __expf(l - mn);
      mrow[j] = mn;
    }
  }
#pragma unroll
  for (int o = 1; o < 16; o <<= 1) {
#pragma unroll
    for (int j = 0; j < 8; ++j) {
      const float mo = __shfl_xor(mrow[j], o, 32);
      const float so = __shfl_xor(srow[j], o, 32);
      const float mn = fmaxf(mrow[j], mo);
      srow[j] = srow[j] * __expf(mrow[j] - mn) + so * __expf(mo - mn);
      mrow[j] = mn;
    }
  }
  float scl[8];
#pragma unroll
  for (int j = 0; j < 8; ++j) scl[j] = 127.f / srow[j];

  v8i pv[8];
#pragma unroll
  for (int dt = 0; dt < 8; ++dt) pv[dt] = v8zero();

  for (int s0 = 0; s0 < T_; s0 += 64) {
#pragma unroll
    for (int st = 0; st < 4; ++st) {
      const int ss = s0 + st * 16;
      const int8_t* kb = K8 + (size_t)(b * T_ + ss + m) * E_ + h * D_ + hi16;
      v8i bk0 = pack8q(*(const uint4*)(kb), *(const uint4*)(kb + 32));
      v8i bk1 = pack8q(*(const uint4*)(kb + 64), *(const uint4*)(kb + 96));
      v8i acc = v8zero();
      acc = wmma_iu8(aQ[0], bk0, acc);
      acc = wmma_iu8(aQ[1], bk1, acc);
      const int scol = ss + m;
#pragma unroll
      for (int j = 0; j < 8; ++j) {
        const int row = r0 + j + 8 * hi;
        const float l = aqk * (float)acc[j] + (scol <= row ? 0.f : -1e4f);
        int p = __float2int_rn(__expf(l - mrow[j]) * scl[j]);
        p = p < 0 ? 0 : (p > 127 ? 127 : p);
        pbuf[(j + 8 * hi) * 64 + st * 16 + m] = (int8_t)p;
      }
    }
    v8i aP = a_frag_lds(pbuf, 0, lane);
#pragma unroll
    for (int dt = 0; dt < 8; ++dt) {
      const int8_t* vb =
          VT8 + ((size_t)(b * H_ + h) * D_ + dt * 16 + m) * T_ + s0 + hi16;
      v8i bv = pack8q(*(const uint4*)(vb), *(const uint4*)(vb + 32));
      pv[dt] = wmma_iu8(aP, bv, pv[dt]);
    }
  }

#pragma unroll
  for (int dt = 0; dt < 8; ++dt) {
    const int col = h * D_ + dt * 16 + m;
#pragma unroll
    for (int j = 0; j < 8; ++j) {
      const int row = r0 + j + 8 * hi;
      CTX8[(size_t)(b * T_ + row) * E_ + col] = sat8(apv * (float)pv[dt][j]);
    }
  }
}

__global__ __launch_bounds__(256) void k_gemm_oproj(
    const int8_t* __restrict__ X, const int8_t* __restrict__ W,
    const float* __restrict__ bias, const float* __restrict__ alpha_p,
    float* __restrict__ out) {
  __shared__ int8_t Als[2][128 * 64];
  __shared__ int8_t Bls[2][128 * 64];
  const int tid = threadIdx.x;
  const int lane = tid & 31, wave = tid >> 5;
  const int wm = wave >> 2, wn = wave & 3;
  const int m0 = blockIdx.y * 128, n0 = blockIdx.x * 128;

  v8i acc[4][2];
#pragma unroll
  for (int i = 0; i < 4; ++i)
#pragma unroll
    for (int j = 0; j < 2; ++j) acc[i][j] = v8zero();

  stage_tile_async(X, W, Als[0], Bls[0], m0, n0, 0, tid);
  int buf = 0;
  for (int kt = 0; kt < E_; kt += 64) {
    if (kt + 64 < E_) {
      stage_tile_async(X, W, Als[buf ^ 1], Bls[buf ^ 1], m0, n0, kt + 64, tid);
      ASYNC_WAIT_4();
    } else {
      ASYNC_WAIT_0();
    }
    __syncthreads();
    v8i bfr0 = b_frag_lds(Bls[buf], wn * 32, lane);
    v8i bfr1 = b_frag_lds(Bls[buf], wn * 32 + 16, lane);
#pragma unroll
    for (int mt = 0; mt < 4; ++mt) {
      v8i afr = a_frag_lds(Als[buf], wm * 64 + mt * 16, lane);
      acc[mt][0] = wmma_iu8(afr, bfr0, acc[mt][0]);
      acc[mt][1] = wmma_iu8(afr, bfr1, acc[mt][1]);
    }
    __syncthreads();
    buf ^= 1;
  }

  const float alpha = *alpha_p;
  const int cl = lane & 15, hi = lane >> 4;
#pragma unroll
  for (int mt = 0; mt < 4; ++mt)
#pragma unroll
    for (int nt = 0; nt < 2; ++nt) {
      const int col = n0 + wn * 32 + nt * 16 + cl;
      const float bv = bias[col];
#pragma unroll
      for (int j = 0; j < 8; ++j) {
        const int row = m0 + wm * 64 + mt * 16 + j + 8 * hi;
        out[(size_t)row * E_ + col] = alpha * (float)acc[mt][nt][j] + bv;
      }
    }
}

extern "C" void kernel_launch(void* const* d_in, const int* in_sizes, int n_in,
                              void* d_out, int out_size, void* d_ws, size_t ws_size,
                              hipStream_t stream) {
  const int* in_hidden = (const int*)d_in[0];
  const int* in_Wq = (const int*)d_in[1];
  const int* in_bq = (const int*)d_in[2];
  const int* in_Wk = (const int*)d_in[3];
  const int* in_bk = (const int*)d_in[4];
  const int* in_Wv = (const int*)d_in[5];
  const int* in_bv = (const int*)d_in[6];
  const int* in_Wo = (const int*)d_in[7];
  const float* b_o = (const float*)d_in[8];
  const float* a_q = (const float*)d_in[10];
  const float* a_k = (const float*)d_in[11];
  const float* a_v = (const float*)d_in[12];
  const float* a_qk = (const float*)d_in[13];
  const float* a_pv = (const float*)d_in[14];
  const float* a_o = (const float*)d_in[15];

  char* ws = (char*)d_ws;
  const size_t MB16 = (size_t)16 << 20;
  int8_t* X8 = (int8_t*)(ws + 0 * MB16);
  int8_t* WQ8 = (int8_t*)(ws + 1 * MB16);
  int8_t* WK8 = (int8_t*)(ws + 2 * MB16);
  int8_t* WV8 = (int8_t*)(ws + 3 * MB16);
  int8_t* WO8 = (int8_t*)(ws + 4 * MB16);
  int8_t* Q8 = (int8_t*)(ws + 5 * MB16);
  int8_t* K8 = (int8_t*)(ws + 6 * MB16);
  int8_t* VT8 = (int8_t*)(ws + 7 * MB16);
  int8_t* CTX8 = (int8_t*)(ws + 8 * MB16);
  int8_t* BQ8 = (int8_t*)(ws + 9 * MB16);
  int8_t* BK8 = BQ8 + E_;
  int8_t* BV8 = BK8 + E_;

  const int n4_big = (B_ * T_ * E_) / 4;
  dim3 pk_blk(256), pk_grid(2048);
  k_pack_i8<<<pk_grid, pk_blk, 0, stream>>>(in_hidden, X8, n4_big);
  k_pack_i8<<<pk_grid, pk_blk, 0, stream>>>(in_Wq, WQ8, n4_big);
  k_pack_i8<<<pk_grid, pk_blk, 0, stream>>>(in_Wk, WK8, n4_big);
  k_pack_i8<<<pk_grid, pk_blk, 0, stream>>>(in_Wv, WV8, n4_big);
  k_pack_i8<<<pk_grid, pk_blk, 0, stream>>>(in_Wo, WO8, n4_big);
  k_pack_i8<<<4, pk_blk, 0, stream>>>(in_bq, BQ8, E_ / 4);
  k_pack_i8<<<4, pk_blk, 0, stream>>>(in_bk, BK8, E_ / 4);
  k_pack_i8<<<4, pk_blk, 0, stream>>>(in_bv, BV8, E_ / 4);

  dim3 g(E_ / 128, (B_ * T_) / 128), blk(256);
  k_gemm_w8a8<<<g, blk, 0, stream>>>(X8, WQ8, BQ8, a_q, Q8, 0);
  k_gemm_w8a8<<<g, blk, 0, stream>>>(X8, WK8, BK8, a_k, K8, 0);
  k_gemm_w8a8<<<g, blk, 0, stream>>>(X8, WV8, BV8, a_v, VT8, 1);

  k_attn<<<dim3(T_ / 128, B_ * H_), blk, 0, stream>>>(Q8, K8, VT8, CTX8, a_qk, a_pv);

  k_gemm_oproj<<<g, blk, 0, stream>>>(CTX8, WO8, b_o, a_o, (float*)d_out);
}
